// BRUCell_57397942943966
// MI455X (gfx1250) — matrix operand order repaired and checked by reading, not compiled, not run on hardware
//
#include <hip/hip_runtime.h>
#include <hip/hip_bf16.h>
#include <stddef.h>
#include <stdint.h>

// Problem constants (fixed by the reference)
#define BB 64
#define TT 512
#define DD 1024
#define UU 1024
#define MM (BB * TT)        // 32768 rows

// GEMM tiling
#define BLK_M 128
#define BLK_N 64
#define BLK_K 32
#define KPAD  40            // LDS row stride (bf16): 80B rows, 16B-aligned, low-conflict

#if __has_builtin(__builtin_amdgcn_global_load_async_to_lds_b128) && \
    __has_builtin(__builtin_amdgcn_s_wait_asynccnt)
#define USE_ASYNC_LDS 1
#else
#define USE_ASYNC_LDS 0
#endif

typedef __bf16 v16bf __attribute__((ext_vector_type(16)));
typedef __bf16 v8bf  __attribute__((ext_vector_type(8)));
typedef float  v8f   __attribute__((ext_vector_type(8)));

__device__ __forceinline__ void split_bf16(float f, __bf16* hi, __bf16* lo) {
    __bf16 h = (__bf16)f;               // RNE to bf16
    *hi = h;
    *lo = (__bf16)(f - (float)h);       // bf16 residual (bf16x3 trick)
}

// 16-element bf16 fragment from two 16-byte LDS chunks.
__device__ __forceinline__ v16bf ld_frag2(const __bf16* p0, const __bf16* p1) {
    union { v16bf v; v8bf h[2]; } u;
    u.h[0] = *(const v8bf*)p0;
    u.h[1] = *(const v8bf*)p1;
    return u.v;
}

#if USE_ASYNC_LDS
// Exact pointee type the builtin expects (per hipcc diagnostic):
// GCC-style 16-byte int vector.
typedef int b128_t __attribute__((vector_size(16)));

// Async 16B global->LDS copy (GLOBAL_LOAD_ASYNC_TO_LDS_B128, ASYNCcnt).
// Address-space pointers built via integer casts, numerically exact on CDNA5:
// flat LDS addr low 32 bits == as(3) address (aperture rule), flat global
// addr == as(1) address.
__device__ __forceinline__ void async_copy16(const void* gsrc, void* ldst) {
    __builtin_amdgcn_global_load_async_to_lds_b128(
        (__attribute__((address_space(1))) b128_t*)(uintptr_t)gsrc,
        (__attribute__((address_space(3))) b128_t*)(uint32_t)(uintptr_t)ldst,
        0, 0);
}
#endif

// ---------------------------------------------------------------------------
// One-time weight prep: W[D][U] fp32 -> Wt_hi/Wt_lo[U][D] bf16 (transposed,
// K-contiguous). Removes 256x-redundant convert+scatter work from the GEMM.
// ---------------------------------------------------------------------------
__global__ __launch_bounds__(256) void w_transpose_split_kernel(
    const float* __restrict__ wz, const float* __restrict__ wr,
    const float* __restrict__ wh,
    __bf16* __restrict__ th, __bf16* __restrict__ tl)
{
    __shared__ float t[32][33];
    const float* W[3] = { wz, wr, wh };
    const float* w = W[blockIdx.z];
    __bf16* oh = th + (size_t)blockIdx.z * UU * DD;
    __bf16* ol = tl + (size_t)blockIdx.z * UU * DD;

    const int u0 = blockIdx.x * 32;
    const int k0 = blockIdx.y * 32;
    const int tx = threadIdx.x;          // 0..31
    const int ty = threadIdx.y;          // 0..7

    #pragma unroll
    for (int i = 0; i < 4; ++i)          // coalesced along U
        t[ty + 8 * i][tx] = w[(size_t)(k0 + ty + 8 * i) * UU + u0 + tx];
    __syncthreads();

    #pragma unroll
    for (int i = 0; i < 4; ++i) {        // coalesced along K
        const int ul = ty + 8 * i;
        const size_t o = (size_t)(u0 + ul) * DD + k0 + tx;
        split_bf16(t[tx][ul], &oh[o], &ol[o]);
    }
}

// ---------------------------------------------------------------------------
// Fused triple GEMM: xz/xr/xh = x @ Wz/Wr/Wh + bias (split-bf16 WMMA).
// Grid (UU/BLK_N, MM/BLK_M) = (16, 256); block 256 = 8 waves.
// A fragments built straight from global fp32 (wave-private rows);
// B tiles double-buffered in LDS via async global->LDS copies (ASYNCcnt).
// ---------------------------------------------------------------------------
__global__ __launch_bounds__(256) void bru_gemm3_kernel(
    const float* __restrict__ x,
    const __bf16* __restrict__ wtH,   // [3][UU][DD] hi plane
    const __bf16* __restrict__ wtL,   // [3][UU][DD] lo plane
    const float* __restrict__ bz, const float* __restrict__ br,
    const float* __restrict__ bh,
    float* __restrict__ xz, float* __restrict__ xr, float* __restrict__ xh)
{
    __shared__ __attribute__((aligned(16))) __bf16 Bst[2][3][2][BLK_N * KPAD]; // 60 KB

    const int tid  = threadIdx.x;
    const int lane = tid & 31;
    const int wv   = tid >> 5;

    const int n0 = blockIdx.x * BLK_N;
    const int m0 = blockIdx.y * BLK_M;

    v8f acc[3][4] = {};   // [matrix][n-tile] 16x16 f32 accumulators

    // A 16x32 layout: lane m = lane&15; K chunks {c8..c8+7} and {16+c8..}, c8=(lane>>4)*8
    const int c8 = (lane >> 4) << 3;
    const float* xrow = x + (size_t)(m0 + (wv << 4) + (lane & 15)) * DD;
    // B 32x16 layout: lane n = lane&15; K chunks {c8..c8+7} and {16+c8..}, c8=(lane>>4)*8
    const int b_k16 = (lane >> 4) << 3;

    // B staging: one 16B chunk (8 bf16) per thread per (mat,plane)
    const int sb_row = (tid * 8) >> 5;   // 0..63
    const int sb_kk  = (tid * 8) & 31;   // 0,8,16,24
    const __bf16* wplane[2] = { wtH, wtL };

#if USE_ASYNC_LDS
    auto stage_B = [&](int k0n, int buf) {
        #pragma unroll
        for (int mat = 0; mat < 3; ++mat)
            #pragma unroll
            for (int pl = 0; pl < 2; ++pl)
                async_copy16(wplane[pl] + (size_t)mat * UU * DD
                                        + (size_t)(n0 + sb_row) * DD + k0n + sb_kk,
                             &Bst[buf][mat][pl][sb_row * KPAD + sb_kk]);
    };
    stage_B(0, 0);
    __builtin_amdgcn_s_wait_asynccnt(0);
#else
    auto load_B = [&](int k0n, uint4* r) {
        #pragma unroll
        for (int mat = 0; mat < 3; ++mat)
            #pragma unroll
            for (int pl = 0; pl < 2; ++pl)
                r[mat * 2 + pl] = *(const uint4*)(wplane[pl] + (size_t)mat * UU * DD
                                   + (size_t)(n0 + sb_row) * DD + k0n + sb_kk);
    };
    auto store_B = [&](int buf, const uint4* r) {
        #pragma unroll
        for (int mat = 0; mat < 3; ++mat)
            #pragma unroll
            for (int pl = 0; pl < 2; ++pl)
                *(uint4*)(&Bst[buf][mat][pl][sb_row * KPAD + sb_kk]) = r[mat * 2 + pl];
    };
    { uint4 r0[6]; load_B(0, r0); store_B(0, r0); }
#endif
    __syncthreads();

    for (int it = 0; it < DD / BLK_K; ++it) {
        const int k0  = it * BLK_K;
        const int buf = it & 1;
        const bool have_next = (it + 1 < DD / BLK_K);

        // ---- issue A loads (global fp32, wave-private rows) ----
        const float4 a0 = *(const float4*)(xrow + k0 + c8);
        const float4 a1 = *(const float4*)(xrow + k0 + c8 + 4);
        const float4 a2 = *(const float4*)(xrow + k0 + 16 + c8);
        const float4 a3 = *(const float4*)(xrow + k0 + 16 + c8 + 4);

        // ---- kick off next B tile into the other buffer ----
#if USE_ASYNC_LDS
        if (have_next) stage_B(k0 + BLK_K, buf ^ 1);
#else
        uint4 rB[6];
        if (have_next) load_B(k0 + BLK_K, rB);
#endif

        // ---- split A to bf16 hi/lo while copies are in flight ----
        union { v16bf v; __bf16 e[16]; } ah, al;
        const float af[16] = { a0.x, a0.y, a0.z, a0.w, a1.x, a1.y, a1.z, a1.w,
                               a2.x, a2.y, a2.z, a2.w, a3.x, a3.y, a3.z, a3.w };
        #pragma unroll
        for (int i = 0; i < 16; ++i) split_bf16(af[i], &ah.e[i], &al.e[i]);

#if !USE_ASYNC_LDS
        if (have_next) store_B(buf ^ 1, rB);
#endif

        // ---- 3 mats x 4 n-tiles x (hi*hi + hi*lo + lo*hi) = 36 WMMAs ----
        #pragma unroll
        for (int mat = 0; mat < 3; ++mat) {
            #pragma unroll
            for (int nt = 0; nt < 4; ++nt) {
                const int boff = ((nt << 4) + (lane & 15)) * KPAD + b_k16;
                const v16bf bhf = ld_frag2(&Bst[buf][mat][0][boff], &Bst[buf][mat][0][boff + 16]);
                const v16bf blf = ld_frag2(&Bst[buf][mat][1][boff], &Bst[buf][mat][1][boff + 16]);
                acc[mat][nt] = __builtin_amdgcn_wmma_f32_16x16x32_bf16(
                    false, ah.v, false, bhf, (short)0, acc[mat][nt], false, false);
                acc[mat][nt] = __builtin_amdgcn_wmma_f32_16x16x32_bf16(
                    false, ah.v, false, blf, (short)0, acc[mat][nt], false, false);
                acc[mat][nt] = __builtin_amdgcn_wmma_f32_16x16x32_bf16(
                    false, al.v, false, bhf, (short)0, acc[mat][nt], false, false);
            }
        }

#if USE_ASYNC_LDS
        if (have_next) __builtin_amdgcn_s_wait_asynccnt(0);  // copies done before barrier
#endif
        __syncthreads();
    }

    // ---- epilogue: bias add + store (C layout: VGPR v -> M=v / v+8) ----
    const float* Bias[3] = { bz, br, bh };
    float*       Outp[3] = { xz, xr, xh };
    const int rowBase = m0 + (wv << 4) + ((lane >> 4) << 3);
    #pragma unroll
    for (int mat = 0; mat < 3; ++mat) {
        float* op = Outp[mat];
        #pragma unroll
        for (int nt = 0; nt < 4; ++nt) {
            const int col = n0 + (nt << 4) + (lane & 15);
            const float bv = Bias[mat][col];
            #pragma unroll
            for (int v = 0; v < 8; ++v) {
                op[(size_t)(rowBase + v) * UU + col] = acc[mat][nt][v] + bv;
            }
        }
    }
}

// ---------------------------------------------------------------------------
// Recurrent scan: memory_z/memory_r act diagonally -> each (b,u) is an
// independent scalar recurrence. One thread per (b,u); fully coalesced.
// ---------------------------------------------------------------------------
__global__ __launch_bounds__(256) void bru_scan_kernel(
    const float* __restrict__ xz, const float* __restrict__ xr,
    const float* __restrict__ xh,
    const float* __restrict__ mz, const float* __restrict__ mr,
    float* __restrict__ out)
{
    const int idx = blockIdx.x * 256 + threadIdx.x;   // 0 .. B*U-1
    const int b = idx >> 10;
    const int u = idx & (UU - 1);
    const float mzu = mz[u];
    const float mru = mr[u];
    const size_t base = (size_t)b * TT * UU + u;

    float h = 0.0f;
    for (int t = 0; t < TT; ++t) {
        const size_t p = base + (size_t)t * UU;
        const float vz = xz[p];
        const float vr = xr[p];
        const float vh = xh[p];
        const float r  = tanhf(vr + h * mru) + 1.0f;
        const float z  = 1.0f / (1.0f + expf(-(vz + h * mzu)));
        const float hh = tanhf(vh + r * h);
        h = (1.0f - z) * hh + z * h;
        out[p] = h;
    }
}

extern "C" void kernel_launch(void* const* d_in, const int* in_sizes, int n_in,
                              void* d_out, int out_size, void* d_ws, size_t ws_size,
                              hipStream_t stream) {
    (void)in_sizes; (void)n_in; (void)out_size; (void)ws_size;

    const float* x  = (const float*)d_in[0];
    const float* wz = (const float*)d_in[1];
    const float* wr = (const float*)d_in[2];
    const float* wh = (const float*)d_in[3];
    const float* mz = (const float*)d_in[4];
    const float* mr = (const float*)d_in[5];
    const float* bz = (const float*)d_in[6];
    const float* br = (const float*)d_in[7];
    const float* bh = (const float*)d_in[8];
    float* out = (float*)d_out;

    const size_t plane = (size_t)MM * UU;          // floats per projection
    float* xz = (float*)d_ws;
    float* xr = xz + plane;
    float* xh = xr + plane;
    __bf16* wtH = (__bf16*)(xh + plane);           // [3][UU][DD] hi
    __bf16* wtL = wtH + (size_t)3 * UU * DD;       // [3][UU][DD] lo

    // 1) one-time weight transpose + bf16 split (12 MB)
    w_transpose_split_kernel<<<dim3(UU / 32, DD / 32, 3), dim3(32, 8), 0, stream>>>(
        wz, wr, wh, wtH, wtL);

    // 2) fused triple GEMM with WMMA
    dim3 gGemm(UU / BLK_N, MM / BLK_M);            // (16, 256): N fastest -> L2 reuse of x
    bru_gemm3_kernel<<<gGemm, 256, 0, stream>>>(x, wtH, wtL, bz, br, bh, xz, xr, xh);

    // 3) diagonal recurrence scan
    const int nScan = BB * UU;                     // 65536 threads
    bru_scan_kernel<<<nScan / 256, 256, 0, stream>>>(xz, xr, xh, mz, mr, out);
}
